// Qwen3MoeBlock_1666447311170
// MI455X (gfx1250) — matrix operand order repaired and checked by reading, not compiled, not run on hardware
//
#include <hip/hip_runtime.h>
#include <hip/hip_bf16.h>

// ---------------------------------------------------------------------------
// Qwen3 MoE block for MI455X (gfx1250, wave32, WMMA bf16 16x16x32,
// async global->LDS staging, double-buffered K pipeline)
// ---------------------------------------------------------------------------
#define T_TOKENS 8192
#define HID      2048
#define NEXP     8
#define INTER    768
#define FC1_N    1536           // 2 * INTER
#define TOPK     2
#define SLOTS_PAD 17408         // 2*T + 8*128 padding headroom
#define MAX_TILES 144

typedef __attribute__((ext_vector_type(16))) __bf16 bf16x16;
typedef __attribute__((ext_vector_type(8)))  __bf16 bf16x8;
typedef __attribute__((ext_vector_type(4)))  __bf16 bf16x4;
typedef __attribute__((ext_vector_type(8)))  float  f32x8;
typedef __attribute__((ext_vector_type(4)))  float  f32x4;

// ---------------------------------------------------------------------------
// CDNA5 async copy: global -> LDS, 128b per lane, tracked by ASYNCcnt.
// dst_lds is the raw LDS byte offset (low 32 bits of a generic LDS pointer,
// per the aperture mapping).
// ---------------------------------------------------------------------------
__device__ __forceinline__ unsigned lds_addr(const void* p) {
  return (unsigned)(unsigned long long)p;
}
__device__ __forceinline__ void async_copy_b128(unsigned dst_lds,
                                                const __bf16* src) {
  asm volatile("global_load_async_to_lds_b128 %0, %1, off"
               :
               : "v"(dst_lds), "v"(src)
               : "memory");
}
__device__ __forceinline__ void wait_async0() {
  asm volatile("s_wait_asynccnt 0x0" ::: "memory");
}

// ---------------------------------------------------------------------------
// WMMA helpers (CDNA5: D = A(16x32 bf16) * B(32x16 bf16) + C(16x16 f32))
// ---------------------------------------------------------------------------
__device__ __forceinline__ f32x8 wmma_bf16(bf16x16 a, bf16x16 b, f32x8 c) {
  return __builtin_amdgcn_wmma_f32_16x16x32_bf16(
      /*neg_a=*/false, a, /*neg_b=*/false, b,
      /*c_mod=*/(short)0, c, /*reuse_a=*/false, /*reuse_b=*/false);
}

// A fragment from LDS tile stored [rows][32] row-major.
// Layout: lane -> row M=(lane&15)+m0, K = kb..kb+15 contiguous, kb=(lane>>4)*16 (matches B frag)
__device__ __forceinline__ bf16x16 load_a_frag(const __bf16* As, int m0, int lane) {
  const __bf16* p = As + (m0 + (lane & 15)) * 32 + ((lane >> 4) * 16);
  bf16x8 lo = *(const bf16x8*)(p);
  bf16x8 hi = *(const bf16x8*)(p + 8);
  return __builtin_shufflevector(lo, hi, 0,1,2,3,4,5,6,7,8,9,10,11,12,13,14,15);
}

// B fragment from LDS tile stored TRANSPOSED [cols][32]: each lane reads
// 32 contiguous bytes: lane -> col N=(lane&15)+n0, K-half = (lane>>4)*16.
__device__ __forceinline__ bf16x16 load_b_frag(const __bf16* Bt, int n0, int lane) {
  return *(const bf16x16*)(Bt + (n0 + (lane & 15)) * 32 + (lane >> 4) * 16);
}

// ---------------------------------------------------------------------------
// 0) init: zero the dense output, mark all slots invalid, zero counters
// ---------------------------------------------------------------------------
__global__ void moe_init_kernel(float* __restrict__ out, int nOut,
                                int* __restrict__ slotTok, int nSlots,
                                int* __restrict__ counts) {
  int i = blockIdx.x * blockDim.x + threadIdx.x;
  int stride = gridDim.x * blockDim.x;
  for (int j = i; j < nOut; j += stride) out[j] = 0.0f;
  for (int j = i; j < nSlots; j += stride) slotTok[j] = -1;
  if (i < NEXP) counts[i] = 0;
}

// ---------------------------------------------------------------------------
// 1) fp32 -> bf16 conversion (4-wide)
// ---------------------------------------------------------------------------
__global__ void cvt_bf16_kernel(const float* __restrict__ src,
                                __bf16* __restrict__ dst, int n4) {
  int i = blockIdx.x * blockDim.x + threadIdx.x;
  int stride = gridDim.x * blockDim.x;
  for (int j = i; j < n4; j += stride) {
    f32x4 v = ((const f32x4*)src)[j];
    ((bf16x4*)dst)[j] = __builtin_convertvector(v, bf16x4);
  }
}

// ---------------------------------------------------------------------------
// 2) router: one wave32 per token. logits -> top-2 -> renormalized probs.
// ---------------------------------------------------------------------------
__global__ __launch_bounds__(256) void router_kernel(
    const float* __restrict__ x, const float* __restrict__ rw,
    float* __restrict__ logits_out, int* __restrict__ topkId,
    float* __restrict__ topkP, int* __restrict__ counts) {
  const int lane = threadIdx.x & 31;
  const int wave = threadIdx.x >> 5;
  const int t = blockIdx.x * 8 + wave;
  const float* xr = x + (size_t)t * HID;

  float acc[NEXP];
#pragma unroll
  for (int e = 0; e < NEXP; ++e) acc[e] = 0.0f;

  for (int h = lane; h < HID; h += 32) {
    float xv = xr[h];
#pragma unroll
    for (int e = 0; e < NEXP; ++e) acc[e] = fmaf(xv, rw[e * HID + h], acc[e]);
  }
#pragma unroll
  for (int e = 0; e < NEXP; ++e) {
#pragma unroll
    for (int s = 16; s > 0; s >>= 1) acc[e] += __shfl_xor(acc[e], s, 32);
  }
  if (lane == 0) {
#pragma unroll
    for (int e = 0; e < NEXP; ++e) logits_out[(size_t)t * NEXP + e] = acc[e];
    int i1 = 0;
#pragma unroll
    for (int e = 1; e < NEXP; ++e) if (acc[e] > acc[i1]) i1 = e;
    int i2 = (i1 == 0) ? 1 : 0;
#pragma unroll
    for (int e = 0; e < NEXP; ++e)
      if (e != i1 && acc[e] > acc[i2]) i2 = e;
    // softmax + top-2 renorm collapses to a 2-way logistic
    float p1 = 1.0f / (1.0f + __expf(acc[i2] - acc[i1]));
    float p2 = 1.0f - p1;
    topkId[t * 2 + 0] = i1; topkId[t * 2 + 1] = i2;
    topkP[t * 2 + 0] = p1;  topkP[t * 2 + 1] = p2;
    atomicAdd(&counts[i1], 1);
    atomicAdd(&counts[i2], 1);
  }
}

// ---------------------------------------------------------------------------
// 3) build padded per-expert offsets and the row-tile worklist
// ---------------------------------------------------------------------------
__global__ void tiles_kernel(const int* __restrict__ counts,
                             int* __restrict__ offs, int* __restrict__ cursor,
                             int* __restrict__ tileE, int* __restrict__ tileBase,
                             int* __restrict__ nTiles) {
  if (threadIdx.x == 0 && blockIdx.x == 0) {
    int off = 0, nt = 0;
    for (int e = 0; e < NEXP; ++e) {
      offs[e] = off;
      cursor[e] = 0;
      int c = counts[e];
      for (int r = 0; r < c; r += 128) {
        tileE[nt] = e;
        tileBase[nt] = off + r;
        ++nt;
      }
      off += ((c + 127) / 128) * 128;
    }
    offs[NEXP] = off;
    *nTiles = nt;
  }
}

// ---------------------------------------------------------------------------
// 4) scatter tokens into expert buckets
// ---------------------------------------------------------------------------
__global__ void scatter_kernel(const int* __restrict__ topkId,
                               const float* __restrict__ topkP,
                               const int* __restrict__ offs,
                               int* __restrict__ cursor,
                               int* __restrict__ slotTok,
                               float* __restrict__ slotP) {
  int t = blockIdx.x * blockDim.x + threadIdx.x;
  if (t >= T_TOKENS) return;
#pragma unroll
  for (int k = 0; k < TOPK; ++k) {
    int e = topkId[t * 2 + k];
    int pos = atomicAdd(&cursor[e], 1);
    int slot = offs[e] + pos;
    slotTok[slot] = t;
    slotP[slot] = topkP[t * 2 + k];
  }
}

// ---------------------------------------------------------------------------
// 5) fc1 grouped GEMM + fused SwiGLU -> bf16 activations
//    tile: 128 gathered rows x (64 gate + 64 up cols), K over 2048 in 32-steps,
//    double-buffered LDS; A staged via global_load_async_to_lds_b128.
//    8 waves = 4(M) x 2(N); per wave 32x32 gate + 32x32 up.
// ---------------------------------------------------------------------------
__global__ __launch_bounds__(256) void fc1_swiglu_kernel(
    const __bf16* __restrict__ xb, const __bf16* __restrict__ w1b,
    const int* __restrict__ slotTok, const int* __restrict__ tileE,
    const int* __restrict__ tileBase, const int* __restrict__ nTiles,
    __bf16* __restrict__ act) {
  __shared__ __align__(64) __bf16 As[2][128 * 32];
  __shared__ __align__(64) __bf16 Bg[2][64 * 32];   // transposed [col][k]
  __shared__ __align__(64) __bf16 Bu[2][64 * 32];   // transposed [col][k]
  __shared__ int toks[128];

  const int tile = blockIdx.x;
  if (tile >= *nTiles) return;
  const int e = tileE[tile];
  const int base = tileBase[tile];
  const int n0 = blockIdx.y * 64;           // gate col tile; up cols at +INTER

  const int tid = threadIdx.x;
  const int lane = tid & 31;
  const int wave = tid >> 5;
  const int wm = (wave >> 1) * 32;
  const int wn = (wave & 1) * 32;

  if (tid < 128) toks[tid] = slotTok[base + tid];
  __syncthreads();

  f32x8 accG[2][2] = {};
  f32x8 accU[2][2] = {};

  const __bf16* w1e = w1b + (size_t)e * HID * FC1_N;

  // staging roles
  const int arow = tid >> 1;                // 0..127
  const int acol = (tid & 1) * 16;          // 0 or 16
  const int atok = toks[arow];
  const int bk = tid >> 3;                  // 0..31 (K row)
  const int bchunk = tid & 7;               // 8 chunks of 16 cols (4 gate + 4 up)
  const int bcol = (bchunk & 3) * 16;
  const bool bup = bchunk >= 4;
  const int bgcol = n0 + bcol + (bup ? INTER : 0);

  const unsigned adst[2] = {lds_addr(&As[0][arow * 32 + acol]),
                            lds_addr(&As[1][arow * 32 + acol])};

  auto stageA = [&](int k0, int buf) {
    if (atok >= 0) {
      const __bf16* s = xb + (size_t)atok * HID + k0 + acol;
      async_copy_b128(adst[buf], s);
      async_copy_b128(adst[buf] + 16, s + 8);
    }
  };
  auto loadB = [&](int k0, bf16x8& o0, bf16x8& o1) {
    const __bf16* s = w1e + (size_t)(k0 + bk) * FC1_N + bgcol;
    o0 = *(const bf16x8*)(s);
    o1 = *(const bf16x8*)(s + 8);
    if (k0 + 32 < HID) __builtin_prefetch(s + (size_t)32 * FC1_N, 0, 3);
  };
  auto storeB = [&](int buf, bf16x8 v0, bf16x8 v1) {
    __bf16* d = bup ? Bu[buf] : Bg[buf];
#pragma unroll
    for (int i = 0; i < 8; ++i) d[(bcol + i) * 32 + bk] = v0[i];
#pragma unroll
    for (int i = 0; i < 8; ++i) d[(bcol + 8 + i) * 32 + bk] = v1[i];
  };

  // ---- prologue: stage k0 = 0 into buffer 0 ----
  {
    stageA(0, 0);
    bf16x8 p0, p1;
    loadB(0, p0, p1);
    storeB(0, p0, p1);
    wait_async0();
    __syncthreads();
  }

  int cur = 0;
  for (int k0 = 0; k0 < HID; k0 += 32) {
    const int nxt = cur ^ 1;
    const bool more = (k0 + 32) < HID;
    bf16x8 nb0 = {}, nb1 = {};
    if (more) {
      stageA(k0 + 32, nxt);         // async global->LDS, overlaps WMMA below
      loadB(k0 + 32, nb0, nb1);     // global loads in flight during compute
    }
    // ---- compute: 8 WMMAs per wave per K-step ----
    bf16x16 af[2];
    af[0] = load_a_frag(As[cur], wm, lane);
    af[1] = load_a_frag(As[cur], wm + 16, lane);
    bf16x16 bg[2], bu[2];
    bg[0] = load_b_frag(Bg[cur], wn, lane);
    bg[1] = load_b_frag(Bg[cur], wn + 16, lane);
    bu[0] = load_b_frag(Bu[cur], wn, lane);
    bu[1] = load_b_frag(Bu[cur], wn + 16, lane);
#pragma unroll
    for (int mi = 0; mi < 2; ++mi) {
#pragma unroll
      for (int ni = 0; ni < 2; ++ni) {
        accG[mi][ni] = wmma_bf16(af[mi], bg[ni], accG[mi][ni]);
        accU[mi][ni] = wmma_bf16(af[mi], bu[ni], accU[mi][ni]);
      }
    }
    if (more) storeB(nxt, nb0, nb1);
    wait_async0();
    __syncthreads();
    cur = nxt;
  }

  // ---- epilogue: silu(gate) * up -> bf16 activation rows ----
  const int hi = lane >> 4;
  const int co = lane & 15;
#pragma unroll
  for (int mi = 0; mi < 2; ++mi) {
#pragma unroll
    for (int ni = 0; ni < 2; ++ni) {
#pragma unroll
      for (int j = 0; j < 8; ++j) {
        int m = wm + mi * 16 + hi * 8 + j;
        if (toks[m] >= 0) {
          float g = accG[mi][ni][j];
          float u = accU[mi][ni][j];
          float s = g / (1.0f + __expf(-g));
          act[(size_t)(base + m) * INTER + n0 + wn + ni * 16 + co] =
              (__bf16)(s * u);
        }
      }
    }
  }
}

// ---------------------------------------------------------------------------
// 6) fc2 grouped GEMM + prob-weighted scatter-add into dense output
//    tile: 128 rows x 128 cols, K over 768, double-buffered, async A staging.
//    8 waves = 4(M) x 2(N of 64).
// ---------------------------------------------------------------------------
__global__ __launch_bounds__(256) void fc2_scatter_kernel(
    const __bf16* __restrict__ act, const __bf16* __restrict__ w2b,
    const int* __restrict__ slotTok, const float* __restrict__ slotP,
    const int* __restrict__ tileE, const int* __restrict__ tileBase,
    const int* __restrict__ nTiles, float* __restrict__ out) {
  __shared__ __align__(64) __bf16 As[2][128 * 32];
  __shared__ __align__(64) __bf16 Bt[2][128 * 32];  // transposed [col][k]
  __shared__ int toks[128];
  __shared__ float probs[128];

  const int tile = blockIdx.x;
  if (tile >= *nTiles) return;
  const int e = tileE[tile];
  const int base = tileBase[tile];
  const int n0 = blockIdx.y * 128;

  const int tid = threadIdx.x;
  const int lane = tid & 31;
  const int wave = tid >> 5;
  const int wm = (wave >> 1) * 32;
  const int wn = (wave & 1) * 64;

  if (tid < 128) {
    toks[tid] = slotTok[base + tid];
    probs[tid] = slotP[base + tid];
  }
  __syncthreads();

  f32x8 acc[2][4] = {};
  const __bf16* w2e = w2b + (size_t)e * INTER * HID;

  const int arow = tid >> 1;
  const int acol = (tid & 1) * 16;
  const int atok = toks[arow];
  const int bk = tid >> 3;
  const int bcol = (tid & 7) * 16;

  const unsigned adst[2] = {lds_addr(&As[0][arow * 32 + acol]),
                            lds_addr(&As[1][arow * 32 + acol])};

  auto stageA = [&](int k0, int buf) {
    if (atok >= 0) {
      const __bf16* s = act + (size_t)(base + arow) * INTER + k0 + acol;
      async_copy_b128(adst[buf], s);
      async_copy_b128(adst[buf] + 16, s + 8);
    }
  };
  auto loadB = [&](int k0, bf16x8& o0, bf16x8& o1) {
    const __bf16* s = w2e + (size_t)(k0 + bk) * HID + n0 + bcol;
    o0 = *(const bf16x8*)(s);
    o1 = *(const bf16x8*)(s + 8);
    if (k0 + 32 < INTER) __builtin_prefetch(s + (size_t)32 * HID, 0, 3);
  };
  auto storeB = [&](int buf, bf16x8 v0, bf16x8 v1) {
    __bf16* d = Bt[buf];
#pragma unroll
    for (int i = 0; i < 8; ++i) d[(bcol + i) * 32 + bk] = v0[i];
#pragma unroll
    for (int i = 0; i < 8; ++i) d[(bcol + 8 + i) * 32 + bk] = v1[i];
  };

  {
    stageA(0, 0);
    bf16x8 p0, p1;
    loadB(0, p0, p1);
    storeB(0, p0, p1);
    wait_async0();
    __syncthreads();
  }

  int cur = 0;
  for (int k0 = 0; k0 < INTER; k0 += 32) {
    const int nxt = cur ^ 1;
    const bool more = (k0 + 32) < INTER;
    bf16x8 nb0 = {}, nb1 = {};
    if (more) {
      stageA(k0 + 32, nxt);
      loadB(k0 + 32, nb0, nb1);
    }
    bf16x16 af[2];
    af[0] = load_a_frag(As[cur], wm, lane);
    af[1] = load_a_frag(As[cur], wm + 16, lane);
    bf16x16 bfr[4];
#pragma unroll
    for (int ni = 0; ni < 4; ++ni)
      bfr[ni] = load_b_frag(Bt[cur], wn + ni * 16, lane);
#pragma unroll
    for (int mi = 0; mi < 2; ++mi) {
#pragma unroll
      for (int ni = 0; ni < 4; ++ni)
        acc[mi][ni] = wmma_bf16(af[mi], bfr[ni], acc[mi][ni]);
    }
    if (more) storeB(nxt, nb0, nb1);
    wait_async0();
    __syncthreads();
    cur = nxt;
  }

  const int hi = lane >> 4;
  const int co = lane & 15;
#pragma unroll
  for (int mi = 0; mi < 2; ++mi) {
#pragma unroll
    for (int j = 0; j < 8; ++j) {
      int m = wm + mi * 16 + hi * 8 + j;
      int t = toks[m];
      if (t >= 0) {
        float p = probs[m];
        float* orow = out + (size_t)t * HID + n0 + wn + co;
#pragma unroll
        for (int ni = 0; ni < 4; ++ni) {
          __hip_atomic_fetch_add(orow + ni * 16, p * acc[mi][ni][j],
                                 __ATOMIC_RELAXED, __HIP_MEMORY_SCOPE_AGENT);
        }
      }
    }
  }
}

// ---------------------------------------------------------------------------
// host-side launcher
// ---------------------------------------------------------------------------
extern "C" void kernel_launch(void* const* d_in, const int* in_sizes, int n_in,
                              void* d_out, int out_size, void* d_ws,
                              size_t ws_size, hipStream_t stream) {
  (void)in_sizes; (void)n_in; (void)out_size; (void)ws_size;
  const float* x  = (const float*)d_in[0];   // (4,2048,2048)
  const float* rw = (const float*)d_in[1];   // (8,2048)
  const float* w1 = (const float*)d_in[2];   // (8,2048,1536)
  const float* w2 = (const float*)d_in[3];   // (8,768,2048)
  float* out = (float*)d_out;                       // (4,2048,2048)
  float* logits = out + (size_t)T_TOKENS * HID;     // (8192,8)

  char* ws = (char*)d_ws;
  size_t off = 0;
  auto alloc = [&](size_t bytes) -> char* {
    char* p = ws + off;
    off = (off + bytes + 255) & ~(size_t)255;
    return p;
  };

  int*    counts   = (int*)alloc(NEXP * 4);
  int*    cursor   = (int*)alloc(NEXP * 4);
  int*    offs     = (int*)alloc((NEXP + 1) * 4);
  int*    nTiles   = (int*)alloc(4);
  int*    tileE    = (int*)alloc(MAX_TILES * 4);
  int*    tileBase = (int*)alloc(MAX_TILES * 4);
  int*    slotTok  = (int*)alloc((size_t)SLOTS_PAD * 4);
  float*  slotP    = (float*)alloc((size_t)SLOTS_PAD * 4);
  int*    topkId   = (int*)alloc((size_t)T_TOKENS * TOPK * 4);
  float*  topkP    = (float*)alloc((size_t)T_TOKENS * TOPK * 4);
  __bf16* xb       = (__bf16*)alloc((size_t)T_TOKENS * HID * 2);
  __bf16* w1b      = (__bf16*)alloc((size_t)NEXP * HID * FC1_N * 2);
  __bf16* w2b      = (__bf16*)alloc((size_t)NEXP * INTER * HID * 2);
  __bf16* act      = (__bf16*)alloc((size_t)SLOTS_PAD * INTER * 2);

  moe_init_kernel<<<2048, 256, 0, stream>>>(out, T_TOKENS * HID, slotTok,
                                            SLOTS_PAD, counts);
  cvt_bf16_kernel<<<2048, 256, 0, stream>>>(x,  xb,  (T_TOKENS * HID) / 4);
  cvt_bf16_kernel<<<2048, 256, 0, stream>>>(w1, w1b, (NEXP * HID * FC1_N) / 4);
  cvt_bf16_kernel<<<2048, 256, 0, stream>>>(w2, w2b, (NEXP * INTER * HID) / 4);
  router_kernel<<<T_TOKENS / 8, 256, 0, stream>>>(x, rw, logits, topkId, topkP,
                                                  counts);
  tiles_kernel<<<1, 32, 0, stream>>>(counts, offs, cursor, tileE, tileBase,
                                     nTiles);
  scatter_kernel<<<T_TOKENS / 256, 256, 0, stream>>>(topkId, topkP, offs,
                                                     cursor, slotTok, slotP);
  fc1_swiglu_kernel<<<dim3(MAX_TILES, INTER / 64), 256, 0, stream>>>(
      xb, w1b, slotTok, tileE, tileBase, nTiles, act);
  fc2_scatter_kernel<<<dim3(MAX_TILES, HID / 128), 256, 0, stream>>>(
      act, w2b, slotTok, slotP, tileE, tileBase, nTiles, out);
}
